// MultiHeadAttention_55087250538766
// MI455X (gfx1250) — matrix operand order repaired and checked by reading, not compiled, not run on hardware
//
#include <hip/hip_runtime.h>

// ---------------------------------------------------------------------------
// MultiHeadAttention for MI455X (gfx1250, wave32, WMMA).
//   x:[1,4096,768] f32, W*:[768,768] f32 (torch layout: y = x @ W^T)
// Pipeline (all bf16 WMMA with f32 accumulation):
//   1) convert x + 4 weights to bf16 (one-shot, negligible traffic)
//   2) Q = x Wq^T -> [H][S][64] bf16 ; K likewise ; V -> transposed [H][64][S]
//   3) flash attention (online softmax; K/V chunks staged in LDS via async
//      global->LDS copies, shared by all 4 waves of the block)
//   4) out = ctx Wfc^T -> f32 [S][768]
// ---------------------------------------------------------------------------

#define SEQ   4096
#define HID   768
#define HEADS 12
#define DH    64
#define SCALE_INV 0.125f   // 1/sqrt(64)

typedef __attribute__((ext_vector_type(16))) __bf16 v16bf;
typedef __attribute__((ext_vector_type(8)))  __bf16 v8bf;
typedef __attribute__((ext_vector_type(4)))  __bf16 v4bf;
typedef __attribute__((ext_vector_type(8)))  float  v8f;
typedef __attribute__((ext_vector_type(4)))  int    v4i;

#if __has_builtin(__builtin_amdgcn_global_load_async_to_lds_b128) && \
    __has_builtin(__builtin_amdgcn_s_wait_asynccnt)
#define HAVE_ASYNC_LDS 1
typedef __attribute__((address_space(1))) v4i* as1_v4i;   // global (prints as __device__)
typedef __attribute__((address_space(3))) v4i* as3_v4i;   // LDS
#endif

// 16-byte global -> LDS copy; async (ASYNCcnt-tracked) when the toolchain has it.
static __device__ inline void copy16B(const __bf16* g, __bf16* l) {
#ifdef HAVE_ASYNC_LDS
  __builtin_amdgcn_global_load_async_to_lds_b128((as1_v4i)g, (as3_v4i)l, 0, 0);
#else
  *(v8bf*)l = *(const v8bf*)g;
#endif
}
static __device__ inline void wait_stage() {
#ifdef HAVE_ASYNC_LDS
  __builtin_amdgcn_s_wait_asynccnt(0);
#endif
}

// ---- WMMA fragment loaders (per CDNA5 ISA §7.12.2 VGPR layouts) ------------
// A (16x32 bf16): lane L holds row M = L%16.
//   lanes 0-15 : VGPR0-3 = K 0..7,  VGPR4-7 = K 16..23
//   lanes16-31 : VGPR0-3 = K 8..15, VGPR4-7 = K 24..31
static __device__ inline v16bf frag_a(const __bf16* row, int k0, int half) {
  v8bf lo = *(const v8bf*)(row + k0 + (half ? 8 : 0));
  v8bf hi = *(const v8bf*)(row + k0 + 16 + (half ? 8 : 0));
  v16bf r;
#pragma unroll
  for (int i = 0; i < 8; ++i) { r[i] = lo[i]; r[8 + i] = hi[i]; }
  return r;
}
// B (32x16 bf16): lane L holds column N = L%16 (a row of the NT operand).
//   lanes 0-15 hold K 0..7, 16..23; lanes 16-31 hold K 8..15, 24..31.
static __device__ inline v16bf frag_b(const __bf16* col, int k0, int half) {
  v8bf lo = *(const v8bf*)(col + k0 + (half ? 8 : 0));
  v8bf hi = *(const v8bf*)(col + k0 + 16 + (half ? 8 : 0));
  v16bf r;
#pragma unroll
  for (int i = 0; i < 8; ++i) { r[i] = lo[i]; r[8 + i] = hi[i]; }
  return r;
}
static __device__ inline v8f wmma_bf16(v16bf a, v16bf b, v8f c) {
  return __builtin_amdgcn_wmma_f32_16x16x32_bf16(
      /*neg_a=*/false, a, /*neg_b=*/false, b,
      /*c_mod=*/(short)0, c, /*reuse_a=*/false, /*reuse_b=*/false);
}

// Butterfly reductions over the 16 lanes that share a C/D-tile row group.
static __device__ inline float redmax16(float v) {
  v = fmaxf(v, __shfl_xor(v, 1));
  v = fmaxf(v, __shfl_xor(v, 2));
  v = fmaxf(v, __shfl_xor(v, 4));
  v = fmaxf(v, __shfl_xor(v, 8));
  return v;
}
static __device__ inline float redsum16(float v) {
  v += __shfl_xor(v, 1);
  v += __shfl_xor(v, 2);
  v += __shfl_xor(v, 4);
  v += __shfl_xor(v, 8);
  return v;
}

// ---- f32 -> bf16 conversion (4 elems / thread) -----------------------------
__global__ void cvt_f32_bf16(const float* __restrict__ in,
                             __bf16* __restrict__ out, int n4) {
  int i = blockIdx.x * blockDim.x + threadIdx.x;
  if (i < n4) {
    float4 f = ((const float4*)in)[i];
    v4bf o;
    o[0] = (__bf16)f.x; o[1] = (__bf16)f.y;
    o[2] = (__bf16)f.z; o[3] = (__bf16)f.w;
    ((v4bf*)out)[i] = o;
  }
}

// ---- NT GEMM: C[M=4096,N=768] = A[4096x768] * B[768x768]^T -----------------
// 256 threads = 8 waves; block tile 128x64; wave tile 32x32 (2x2 WMMA tiles).
// mode 0: store bf16 headed   [H][S][DH]   (Q, K)
// mode 1: store bf16 transposed [H][DH][S] (V)
// mode 2: store f32 row-major [S][HID]     (final FC output)
__global__ __launch_bounds__(256) void gemm_nt(const __bf16* __restrict__ A,
                                               const __bf16* __restrict__ B,
                                               void* __restrict__ C, int mode) {
  const int K = HID, N = HID;
  const int lane = threadIdx.x & 31, wave = threadIdx.x >> 5;
  const int lane16 = lane & 15, half = lane >> 4;
  const int m0 = blockIdx.x * 128 + (wave & 3) * 32;
  const int n0 = blockIdx.y * 64 + (wave >> 2) * 32;

  const __bf16* arow0 = A + (size_t)(m0 + lane16) * K;
  const __bf16* arow1 = arow0 + (size_t)16 * K;
  const __bf16* brow0 = B + (size_t)(n0 + lane16) * K;
  const __bf16* brow1 = brow0 + (size_t)16 * K;

  v8f acc[2][2] = {};
  for (int k0 = 0; k0 < K; k0 += 32) {
    // locality 3 -> near-cache prefetch (data is L2-resident; pull into WGP$)
    __builtin_prefetch(arow0 + k0 + 128, 0, 3);
    __builtin_prefetch(brow0 + k0 + 128, 0, 3);
    v16bf a0 = frag_a(arow0, k0, half);
    v16bf a1 = frag_a(arow1, k0, half);
    v16bf b0 = frag_b(brow0, k0, half);
    v16bf b1 = frag_b(brow1, k0, half);
    acc[0][0] = wmma_bf16(a0, b0, acc[0][0]);
    acc[0][1] = wmma_bf16(a0, b1, acc[0][1]);
    acc[1][0] = wmma_bf16(a1, b0, acc[1][0]);
    acc[1][1] = wmma_bf16(a1, b1, acc[1][1]);
  }

  // D layout: VGPR r -> row (r + 8*half), column lane16 of each 16x16 tile.
#pragma unroll
  for (int mt = 0; mt < 2; ++mt)
#pragma unroll
    for (int nt = 0; nt < 2; ++nt)
#pragma unroll
      for (int r = 0; r < 8; ++r) {
        int m = m0 + mt * 16 + r + 8 * half;
        int n = n0 + nt * 16 + lane16;
        float v = acc[mt][nt][r];
        if (mode == 2) {
          ((float*)C)[(size_t)m * N + n] = v;
        } else {
          int h = n >> 6, dh = n & 63;
          __bf16 bv = (__bf16)v;
          if (mode == 0)
            ((__bf16*)C)[((size_t)h * SEQ + m) * DH + dh] = bv;     // [H][S][DH]
          else
            ((__bf16*)C)[((size_t)h * DH + dh) * SEQ + m] = bv;     // [H][DH][S]
        }
      }
}

// ---- Flash attention: grid (SEQ/64, HEADS), 128 threads = 4 waves ----------
// Each wave owns 16 query rows. Per 32-key chunk the block stages K (32x64)
// and V^T (64x32) once into LDS (async global->LDS) — all 4 waves share them,
// cutting global fragment traffic 4x. Scores/O stay in D-layout registers;
// P transposes to A-layout via a per-wave LDS buffer.
__global__ __launch_bounds__(128) void attn(const __bf16* __restrict__ Q,
                                            const __bf16* __restrict__ Kk,
                                            const __bf16* __restrict__ Vt,
                                            __bf16* __restrict__ ctx) {
  const int h = blockIdx.y;
  const int tid = threadIdx.x;
  const int lane = tid & 31, wave = tid >> 5;
  const int lane16 = lane & 15, half = lane >> 4;
  const int q0 = blockIdx.x * 64 + wave * 16;

  __shared__ __align__(16) __bf16 kbuf[32 * DH];      // [key][d]   4 KB
  __shared__ __align__(16) __bf16 vbuf[DH * 32];      // [d][key]   4 KB
  __shared__ __align__(16) __bf16 pbuf[4][16 * 32];   // per-wave P 4 KB

  const __bf16* qrow = Q + ((size_t)h * SEQ + q0 + lane16) * DH;
  v16bf qa0 = frag_a(qrow, 0, half);
  v16bf qa1 = frag_a(qrow, 32, half);

  const __bf16* kbase = Kk + (size_t)h * SEQ * DH;
  const __bf16* vbase = Vt + (size_t)h * DH * SEQ;
  __bf16* pw = pbuf[wave];

  v8f acc[4] = {};
  float rowm[8], rowl[8];
#pragma unroll
  for (int r = 0; r < 8; ++r) { rowm[r] = -1e30f; rowl[r] = 0.0f; }

  for (int kv = 0; kv < SEQ; kv += 32) {
    // --- stage K chunk (32 rows x 128B) and V^T chunk (64 rows x 64B) ------
    // 256 + 256 sixteen-byte copies, 128 threads -> 2 each per buffer.
    for (int c = tid; c < 256; c += 128) {
      copy16B(kbase + (size_t)(kv + (c >> 3)) * DH + (c & 7) * 8, kbuf + c * 8);
      copy16B(vbase + (size_t)(c >> 2) * SEQ + kv + (c & 3) * 8, vbuf + c * 8);
    }
    wait_stage();
    __syncthreads();

    // --- scores S = (Q K^T) * scale for 16 rows x 32 keys (from LDS) -------
    const __bf16* k0p = kbuf + (size_t)lane16 * DH;
    const __bf16* k1p = kbuf + (size_t)(16 + lane16) * DH;
    v8f s0 = {}, s1 = {};
    s0 = wmma_bf16(qa0, frag_b(k0p, 0, half), s0);
    s0 = wmma_bf16(qa1, frag_b(k0p, 32, half), s0);
    s1 = wmma_bf16(qa0, frag_b(k1p, 0, half), s1);
    s1 = wmma_bf16(qa1, frag_b(k1p, 32, half), s1);

    // --- online softmax (rows r+8*half live in this lane's half-group) -----
#pragma unroll
    for (int r = 0; r < 8; ++r) {
      float a = s0[r] * SCALE_INV;
      float b = s1[r] * SCALE_INV;
      float mnew = fmaxf(rowm[r], redmax16(fmaxf(a, b)));
      float corr = __expf(rowm[r] - mnew);
      float e0 = __expf(a - mnew);
      float e1 = __expf(b - mnew);
      rowl[r] = rowl[r] * corr + redsum16(e0 + e1);
      rowm[r] = mnew;
#pragma unroll
      for (int dt = 0; dt < 4; ++dt) acc[dt][r] *= corr;
      int m = r + 8 * half;                    // D-layout row -> LDS row-major
      pw[m * 32 + lane16] = (__bf16)e0;
      pw[m * 32 + 16 + lane16] = (__bf16)e1;
    }
    __syncthreads();

    // --- P (16x32) back as an A-fragment, then O += P * V (V from LDS) -----
    v16bf pa = frag_a(pw + lane16 * 32, 0, half);
#pragma unroll
    for (int dt = 0; dt < 4; ++dt) {
      const __bf16* vcol = vbuf + (size_t)(dt * 16 + lane16) * 32;
      acc[dt] = wmma_bf16(pa, frag_b(vcol, 0, half), acc[dt]);
    }
    __syncthreads();   // protect kbuf/vbuf/pbuf before next chunk's staging
  }

  // --- normalize and emit ctx in [S][HID] bf16 (feeds the FC NT GEMM) ------
#pragma unroll
  for (int dt = 0; dt < 4; ++dt)
#pragma unroll
    for (int r = 0; r < 8; ++r) {
      int srow = q0 + r + 8 * half;
      int e = h * DH + dt * 16 + lane16;
      ctx[(size_t)srow * HID + e] = (__bf16)(acc[dt][r] / rowl[r]);
    }
}

// ---------------------------------------------------------------------------
extern "C" void kernel_launch(void* const* d_in, const int* in_sizes, int n_in,
                              void* d_out, int out_size, void* d_ws, size_t ws_size,
                              hipStream_t stream) {
  (void)in_sizes; (void)n_in; (void)out_size; (void)ws_size;
  const float* x   = (const float*)d_in[0];
  const float* wq  = (const float*)d_in[1];
  const float* wk  = (const float*)d_in[2];
  const float* wv  = (const float*)d_in[3];
  const float* wfc = (const float*)d_in[4];
  float* out = (float*)d_out;

  const size_t nx = (size_t)SEQ * HID;   // 3,145,728
  const size_t nw = (size_t)HID * HID;   //   589,824
  __bf16* xb   = (__bf16*)d_ws;
  __bf16* wqb  = xb + nx;
  __bf16* wkb  = wqb + nw;
  __bf16* wvb  = wkb + nw;
  __bf16* wfcb = wvb + nw;
  __bf16* qb   = wfcb + nw;              // [H][S][DH]
  __bf16* kb   = qb + nx;                // [H][S][DH]
  __bf16* vtb  = kb + nx;                // [H][DH][S]
  __bf16* ctxb = vtb + nx;               // [S][HID]
  // total scratch: (5*nx + 4*nw) * 2B  ~= 36.2 MB

  const int T = 256;
  cvt_f32_bf16<<<(int)((nx / 4 + T - 1) / T), T, 0, stream>>>(x,   xb,   (int)(nx / 4));
  cvt_f32_bf16<<<(int)((nw / 4 + T - 1) / T), T, 0, stream>>>(wq,  wqb,  (int)(nw / 4));
  cvt_f32_bf16<<<(int)((nw / 4 + T - 1) / T), T, 0, stream>>>(wk,  wkb,  (int)(nw / 4));
  cvt_f32_bf16<<<(int)((nw / 4 + T - 1) / T), T, 0, stream>>>(wv,  wvb,  (int)(nw / 4));
  cvt_f32_bf16<<<(int)((nw / 4 + T - 1) / T), T, 0, stream>>>(wfc, wfcb, (int)(nw / 4));

  dim3 gg(SEQ / 128, HID / 64);          // 32 x 12 blocks
  gemm_nt<<<gg, 256, 0, stream>>>(xb, wqb, qb,  0);
  gemm_nt<<<gg, 256, 0, stream>>>(xb, wkb, kb,  0);
  gemm_nt<<<gg, 256, 0, stream>>>(xb, wvb, vtb, 1);

  dim3 ga(SEQ / 64, HEADS);              // 64 x 12 blocks, 4 waves each
  attn<<<ga, 128, 0, stream>>>(qb, kb, vtb, ctxb);

  gemm_nt<<<gg, 256, 0, stream>>>(ctxb, wfcb, out, 2);
}
